// DNN_70695161692570
// MI455X (gfx1250) — compile-verified, matrix operand order checked by reading, not run on hardware
//
#include <hip/hip_runtime.h>
#include <hip/hip_bf16.h>

typedef __attribute__((ext_vector_type(16))) _Float16 v16h;
typedef __attribute__((ext_vector_type(8)))  _Float16 v8h;
typedef __attribute__((ext_vector_type(8)))  float    v8f;
typedef __attribute__((ext_vector_type(4)))  unsigned int u32x4;
typedef __attribute__((ext_vector_type(8)))  int i32x8;
typedef __attribute__((ext_vector_type(4)))  int i32x4;

#define N_TOK 6400   // B*L
#define DIM   64
#define BATCH 128
#define SEQL  50
#define KCB   1024
#define NITER (N_TOK / 32)

union FragU { v16h v; v8h h[2]; };

// Loads one 16x32 f16 WMMA operand fragment (A-layout, or B via its row-major
// transpose) from a row-major matrix. Per CDNA5 ISA 16-bit A layout:
//   lanes 0-15:  row = lane,    halves 0-7 = k[0..7],  halves 8-15 = k[16..23]
//   lanes 16-31: row = lane-16, halves 0-7 = k[8..15], halves 8-15 = k[24..31]
__device__ __forceinline__ v16h frag_load(const _Float16* __restrict__ base, int ld) {
  int lane = threadIdx.x & 31;
  int r    = lane & 15;
  int koff = (lane >> 4) << 3;          // 0 or 8
  const _Float16* p = base + r * ld + koff;
  FragU f;
  f.h[0] = *(const v8h*)(p);            // 16B -> global/ds load_b128
  f.h[1] = *(const v8h*)(p + 16);
  return f.v;
}

__device__ __forceinline__ v8f wmma_f16(v16h a, v16h b, v8f c) {
  return __builtin_amdgcn_wmma_f32_16x16x32_f16(false, a, false, b, (short)0, c, false, false);
}

// ---- Tensor Data Mover: 2D tile global->LDS (D# per ISA §8.3/8.4) ----
// data_size = 8B units. Groups 2/3 zero (2D tile, tile_dim2=0).
union G0U { u32x4 v; unsigned int w[4]; };
union G1U { i32x8 v; unsigned short h[16]; };

__device__ __forceinline__ void tdm_load_2d(const void* gaddr, unsigned int lds_addr,
                                            unsigned int tile_d0, unsigned int tile_d1,
                                            unsigned int tensor_d0, unsigned int tensor_d1,
                                            unsigned long long stride0_u8) {
  G0U g0; G1U g1;
  unsigned long long ga = (unsigned long long)(uintptr_t)gaddr;
  g0.w[0] = 1u;                                                   // count=1, user mode
  g0.w[1] = lds_addr;                                             // LDS byte address
  g0.w[2] = (unsigned int)(ga & 0xFFFFFFFFu);                     // global_addr[31:0]
  g0.w[3] = (unsigned int)((ga >> 32) & 0x1FFFFFFu) | (2u << 30); // addr[56:32] | type=2
  #pragma unroll
  for (int i = 0; i < 16; ++i) g1.h[i] = 0;
  g1.h[1]  = 3;                                          // data_size = 8 bytes
  g1.h[3]  = (unsigned short)(tensor_d0 & 0xFFFF);       // tensor_dim0 [79:48]
  g1.h[4]  = (unsigned short)(tensor_d0 >> 16);
  g1.h[5]  = (unsigned short)(tensor_d1 & 0xFFFF);       // tensor_dim1 [111:80]
  g1.h[6]  = (unsigned short)(tensor_d1 >> 16);
  g1.h[7]  = (unsigned short)tile_d0;                    // tile_dim0 [127:112]
  g1.h[8]  = (unsigned short)tile_d1;                    // tile_dim1 [143:128]
  g1.h[10] = (unsigned short)(stride0_u8 & 0xFFFF);      // tensor_dim0_stride [207:160]
  g1.h[11] = (unsigned short)((stride0_u8 >> 16) & 0xFFFF);
  g1.h[12] = (unsigned short)((stride0_u8 >> 32) & 0xFFFF);
  i32x4 z4 = {0, 0, 0, 0};
  i32x8 z8 = {0, 0, 0, 0, 0, 0, 0, 0};
  __builtin_amdgcn_tensor_load_to_lds(g0.v, g1.v, z4, z4, z8, 0);
}

__device__ __forceinline__ unsigned int lds_off(const void* p) {
  // LDS generic addresses carry the workgroup-relative byte offset in [31:0]
  return (unsigned int)(uintptr_t)p;
}

// ---------------- K1: embedding gather + mask, fp32 + f16 copies ----------------
__global__ void gather_kernel(const int* __restrict__ ids, const int* __restrict__ masks,
                              const float* __restrict__ emb,
                              float* __restrict__ h32, _Float16* __restrict__ h16) {
  int i = blockIdx.x * blockDim.x + threadIdx.x;
  if (i >= N_TOK * DIM) return;
  int t = i >> 6, d = i & 63;
  float m = (masks[t] >= 1) ? 1.0f : 0.0f;
  float val = emb[ids[t] * DIM + d] * m;
  h32[i] = val;
  h16[i] = (_Float16)val;
}

// ---------------- K2a: transpose+f16 the three 64x64 projection matrices ----------------
__global__ void prep_w_kernel(const float* __restrict__ Wq, const float* __restrict__ Wk,
                              const float* __restrict__ Wv, _Float16* __restrict__ wt16) {
  int i = blockIdx.x * blockDim.x + threadIdx.x;  // 3*64*64
  if (i >= 3 * DIM * DIM) return;
  int mat = i / (DIM * DIM);
  int rem = i % (DIM * DIM);
  int n = rem >> 6, d = rem & 63;
  const float* W = (mat == 0) ? Wq : (mat == 1) ? Wk : Wv;
  wt16[i] = (_Float16)W[d * DIM + n];   // wt[mat][n][d] = W[d][n]
}

// ---------------- K2b: codebook -> f16 + half squared norms ----------------
__global__ void prep_cb_kernel(const float* __restrict__ cb,
                               _Float16* __restrict__ cb16, float* __restrict__ cn) {
  int j = blockIdx.x * blockDim.x + threadIdx.x;
  if (j >= KCB) return;
  float s = 0.0f;
  for (int d = 0; d < DIM; ++d) {
    float v = cb[j * DIM + d];
    s += v * v;
    cb16[j * DIM + d] = (_Float16)v;
  }
  cn[j] = 0.5f * s;
}

// ---------------- K3: QKV projection via WMMA; V stored transposed ----------------
__global__ __launch_bounds__(256) void qkv_kernel(const _Float16* __restrict__ h16,
                                                  const _Float16* __restrict__ wt16,
                                                  const float* __restrict__ bq,
                                                  const float* __restrict__ bk,
                                                  const float* __restrict__ bv,
                                                  _Float16* __restrict__ q16,
                                                  _Float16* __restrict__ k16,
                                                  _Float16* __restrict__ vt16) {
  int w = threadIdx.x >> 5, lane = threadIdx.x & 31;
  int rt = blockIdx.x * 8 + w;               // 16-row tile, 0..399
  int half = lane >> 4, nl = lane & 15;

  v16h ha0 = frag_load(h16 + rt * 16 * DIM, DIM);        // d 0..31
  v16h ha1 = frag_load(h16 + rt * 16 * DIM + 32, DIM);   // d 32..63

  #pragma unroll
  for (int mat = 0; mat < 3; ++mat) {
    const _Float16* wt = wt16 + mat * DIM * DIM;
    const float* bias = (mat == 0) ? bq : (mat == 1) ? bk : bv;
    #pragma unroll
    for (int nt = 0; nt < 4; ++nt) {
      float bb = bias[nt * 16 + nl];
      v8f acc = {bb, bb, bb, bb, bb, bb, bb, bb};
      v16h wb0 = frag_load(wt + nt * 16 * DIM, DIM);
      v16h wb1 = frag_load(wt + nt * 16 * DIM + 32, DIM);
      acc = wmma_f16(ha0, wb0, acc);
      acc = wmma_f16(ha1, wb1, acc);
      int n = nt * 16 + nl;
      if (mat < 2) {
        #pragma unroll
        for (int r = 0; r < 8; ++r) {
          int row = rt * 16 + r + 8 * half;
          _Float16 hv = (_Float16)acc[r];
          if (mat == 0) q16[row * DIM + n] = hv;
          else          k16[row * DIM + n] = hv;
        }
      } else {
        // V^T store: consecutive r are consecutive rows -> one b128 store
        v8h pk;
        #pragma unroll
        for (int r = 0; r < 8; ++r) pk[r] = (_Float16)acc[r];
        *(v8h*)(vt16 + n * N_TOK + rt * 16 + 8 * half) = pk;
      }
    }
  }
}

// ---------------- K4: flash attention, TDM-staged K/V^T tiles in LDS ----------------
__global__ __launch_bounds__(256) void attn_kernel(const _Float16* __restrict__ q16,
                                                   const _Float16* __restrict__ k16,
                                                   const _Float16* __restrict__ vt16,
                                                   float* __restrict__ ha32,
                                                   _Float16* __restrict__ ha16) {
  __shared__ _Float16 kbuf[2][32][64];      // 2 x 4KB K tile (row-major)
  __shared__ _Float16 vbuf[2][64][32];      // 2 x 4KB V^T tile (row-major, ld=32)
  __shared__ _Float16 psh[8][16][32];       // per-wave P tile
  int w = threadIdx.x >> 5, lane = threadIdx.x & 31;
  int qt = blockIdx.x * 8 + w;              // query tile 0..399
  int half = lane >> 4, nl = lane & 15;

  v16h qa0 = frag_load(q16 + qt * 16 * DIM, DIM);
  v16h qa1 = frag_load(q16 + qt * 16 * DIM + 32, DIM);

  float mprev[8], lsum[8];
  v8f o0 = {}, o1 = {}, o2 = {}, o3 = {};
  #pragma unroll
  for (int r = 0; r < 8; ++r) { mprev[r] = -__builtin_inff(); lsum[r] = 0.0f; }

  // Prologue: stage tiles for kt=0.  K tile = contiguous 4KB (512 x 8B);
  // V^T tile = 64 rows x 64B, global row stride 12800B = 1600 x 8B.
  if (w == 0) {
    tdm_load_2d(k16, lds_off(&kbuf[0][0][0]), 512, 1, N_TOK * DIM / 4, 1, N_TOK * DIM / 4);
    tdm_load_2d(vt16, lds_off(&vbuf[0][0][0]), 8, 64, N_TOK / 4, 64, N_TOK / 4);
  }

  for (int kt = 0; kt < NITER; ++kt) {
    int buf = kt & 1;
    if (w == 0) {
      if (kt + 1 < NITER) {
        int nb = buf ^ 1;
        int kb2 = (kt + 1) * 32;
        tdm_load_2d(k16 + kb2 * DIM, lds_off(&kbuf[nb][0][0]),
                    512, 1, N_TOK * DIM / 4, 1, N_TOK * DIM / 4);
        tdm_load_2d(vt16 + kb2, lds_off(&vbuf[nb][0][0]),
                    8, 64, N_TOK / 4, 64, N_TOK / 4);
        __builtin_amdgcn_s_wait_tensorcnt(2);   // tiles for kt are complete
      } else {
        __builtin_amdgcn_s_wait_tensorcnt(0);
      }
    }
    __syncthreads();                            // tiles visible to all 8 waves

    v8f zero = {};
    // scores for 2 x 16-key subtiles, contraction over D=64 in two K=32 chunks
    v16h kf0 = frag_load(&kbuf[buf][0][0], DIM);
    v16h kf1 = frag_load(&kbuf[buf][0][0] + 32, DIM);
    v8f s0 = wmma_f16(qa0, kf0, zero);
    s0 = wmma_f16(qa1, kf1, s0);
    v16h kf2 = frag_load(&kbuf[buf][16][0], DIM);
    v16h kf3 = frag_load(&kbuf[buf][16][0] + 32, DIM);
    v8f s1 = wmma_f16(qa0, kf2, zero);
    s1 = wmma_f16(qa1, kf3, s1);

    // online softmax in C-layout (col = lane within half, row = vgpr idx)
    float p0[8], p1[8];
    #pragma unroll
    for (int r = 0; r < 8; ++r) {
      float a = s0[r] * 0.125f;             // 1/sqrt(64)
      float b = s1[r] * 0.125f;
      float mx = fmaxf(a, b);
      #pragma unroll
      for (int off = 1; off < 16; off <<= 1)
        mx = fmaxf(mx, __shfl_xor(mx, off, 32));
      float mnew = fmaxf(mprev[r], mx);
      float alpha = __expf(mprev[r] - mnew);
      float e0 = __expf(a - mnew), e1 = __expf(b - mnew);
      float rs = e0 + e1;
      #pragma unroll
      for (int off = 1; off < 16; off <<= 1)
        rs += __shfl_xor(rs, off, 32);
      lsum[r] = lsum[r] * alpha + rs;
      mprev[r] = mnew;
      p0[r] = e0; p1[r] = e1;
      o0[r] = o0[r] * alpha;
      o1[r] = o1[r] * alpha;
      o2[r] = o2[r] * alpha;
      o3[r] = o3[r] * alpha;
    }
    // C-layout -> A-layout transpose of P through LDS (in-order DS within wave)
    #pragma unroll
    for (int r = 0; r < 8; ++r) {
      int m = r + 8 * half;
      psh[w][m][nl]      = (_Float16)p0[r];
      psh[w][m][16 + nl] = (_Float16)p1[r];
    }
    asm volatile("s_wait_dscnt 0" ::: "memory");
    v16h pa = frag_load(&psh[w][0][0], 32);
    asm volatile("" ::: "memory");
    // O += P @ V  (B operand = V^T tile in LDS, ld=32)
    {
      v16h vb = frag_load(&vbuf[buf][0][0], 32);
      o0 = wmma_f16(pa, vb, o0);
    }
    {
      v16h vb = frag_load(&vbuf[buf][16][0], 32);
      o1 = wmma_f16(pa, vb, o1);
    }
    {
      v16h vb = frag_load(&vbuf[buf][32][0], 32);
      o2 = wmma_f16(pa, vb, o2);
    }
    {
      v16h vb = frag_load(&vbuf[buf][48][0], 32);
      o3 = wmma_f16(pa, vb, o3);
    }
    asm volatile("s_wait_dscnt 0" ::: "memory");  // DS reads done before refill
    __syncthreads();
  }

  #pragma unroll
  for (int r = 0; r < 8; ++r) {
    int row = qt * 16 + r + 8 * half;
    float inv = 1.0f / lsum[r];
    float v0 = o0[r] * inv, v1 = o1[r] * inv, v2 = o2[r] * inv, v3 = o3[r] * inv;
    ha32[row * DIM + nl]      = v0;  ha16[row * DIM + nl]      = (_Float16)v0;
    ha32[row * DIM + 16 + nl] = v1;  ha16[row * DIM + 16 + nl] = (_Float16)v1;
    ha32[row * DIM + 32 + nl] = v2;  ha16[row * DIM + 32 + nl] = (_Float16)v2;
    ha32[row * DIM + 48 + nl] = v3;  ha16[row * DIM + 48 + nl] = (_Float16)v3;
  }
}

// ---------------- K5: VQ argmin via WMMA (argmax of f.c - 0.5|c|^2) ----------------
__global__ __launch_bounds__(256) void vq_kernel(const _Float16* __restrict__ ha16,
                                                 const _Float16* __restrict__ cb16,
                                                 const float* __restrict__ cn,
                                                 int* __restrict__ idx_out) {
  int w = threadIdx.x >> 5, lane = threadIdx.x & 31;
  int qt = blockIdx.x * 8 + w;
  int half = lane >> 4, nl = lane & 15;

  v16h fa0 = frag_load(ha16 + qt * 16 * DIM, DIM);
  v16h fa1 = frag_load(ha16 + qt * 16 * DIM + 32, DIM);

  float best[8]; int bidx[8];
  #pragma unroll
  for (int r = 0; r < 8; ++r) { best[r] = -__builtin_inff(); bidx[r] = 0; }

  for (int ct = 0; ct < KCB / 16; ++ct) {
    v8f z = {};
    v16h cf0 = frag_load(cb16 + ct * 16 * DIM, DIM);
    v16h cf1 = frag_load(cb16 + ct * 16 * DIM + 32, DIM);
    v8f g = wmma_f16(fa0, cf0, z);
    g = wmma_f16(fa1, cf1, g);
    int n = ct * 16 + nl;
    float hn = cn[n];
    #pragma unroll
    for (int r = 0; r < 8; ++r) {
      float val = g[r] - hn;
      if (val > best[r]) { best[r] = val; bidx[r] = n; }
    }
  }
  #pragma unroll
  for (int r = 0; r < 8; ++r) {
    #pragma unroll
    for (int off = 1; off < 16; off <<= 1) {
      float ov = __shfl_xor(best[r], off, 32);
      int   oi = __shfl_xor(bidx[r], off, 32);
      if (ov > best[r] || (ov == best[r] && oi < bidx[r])) { best[r] = ov; bidx[r] = oi; }
    }
  }
  if (nl == 0) {
    #pragma unroll
    for (int r = 0; r < 8; ++r)
      idx_out[qt * 16 + r + 8 * half] = bidx[r];
  }
}

// ---------------- K6: means + concat + final 128x128x64 GEMM ----------------
__global__ __launch_bounds__(64) void final_kernel(const float* __restrict__ ha32,
                                                   const int* __restrict__ idx,
                                                   const int* __restrict__ masks,
                                                   const float* __restrict__ cb,
                                                   const float* __restrict__ W_enc,
                                                   const float* __restrict__ b_enc,
                                                   float* __restrict__ out) {
  __shared__ float x[2 * DIM];
  int b = blockIdx.x, d = threadIdx.x;
  float denom = 0.0f;
  for (int l = 0; l < SEQL; ++l)
    denom += (masks[b * SEQL + l] >= 1) ? 1.0f : 0.0f;
  float hs = 0.0f, vs = 0.0f;
  for (int l = 0; l < SEQL; ++l) {
    int row = b * SEQL + l;
    hs += ha32[row * DIM + d];
    vs += cb[idx[row] * DIM + d];
  }
  x[d]       = vs / denom;            // vq_mean: reference has no epsilon here
  x[DIM + d] = hs / (denom + 1e-9f);  // hist_mean: +1e-9 per reference
  __syncthreads();
  float acc = b_enc[d];
  for (int i = 0; i < 2 * DIM; ++i)
    acc += x[i] * W_enc[i * DIM + d];
  out[b * DIM + d] = acc;
}

extern "C" void kernel_launch(void* const* d_in, const int* in_sizes, int n_in,
                              void* d_out, int out_size, void* d_ws, size_t ws_size,
                              hipStream_t stream) {
  const int*   ids   = (const int*)d_in[0];
  const int*   masks = (const int*)d_in[1];
  const float* emb   = (const float*)d_in[2];
  const float* cb    = (const float*)d_in[3];
  const float* Wq    = (const float*)d_in[4];
  const float* bq    = (const float*)d_in[5];
  const float* Wk    = (const float*)d_in[6];
  const float* bk    = (const float*)d_in[7];
  const float* Wv    = (const float*)d_in[8];
  const float* bv    = (const float*)d_in[9];
  const float* W_enc = (const float*)d_in[10];
  const float* b_enc = (const float*)d_in[11];
  float* out = (float*)d_out;

  char* ws = (char*)d_ws;
  size_t off = 0;
  auto A = [&](size_t bytes) -> char* {
    char* p = ws + off;
    off += (bytes + 255) & ~(size_t)255;
    return p;
  };
  float*    h32  = (float*)   A((size_t)N_TOK * DIM * 4);
  _Float16* h16  = (_Float16*)A((size_t)N_TOK * DIM * 2);
  _Float16* q16  = (_Float16*)A((size_t)N_TOK * DIM * 2);
  _Float16* k16  = (_Float16*)A((size_t)N_TOK * DIM * 2);
  _Float16* vt16 = (_Float16*)A((size_t)N_TOK * DIM * 2);
  _Float16* wt16 = (_Float16*)A((size_t)3 * DIM * DIM * 2);
  float*    ha32 = (float*)   A((size_t)N_TOK * DIM * 4);
  _Float16* ha16 = (_Float16*)A((size_t)N_TOK * DIM * 2);
  _Float16* cb16 = (_Float16*)A((size_t)KCB * DIM * 2);
  float*    cn   = (float*)   A((size_t)KCB * 4);
  int*      vidx = (int*)     A((size_t)N_TOK * 4);
  (void)h32; (void)ws_size; (void)in_sizes; (void)n_in; (void)out_size;

  gather_kernel<<<(N_TOK * DIM + 255) / 256, 256, 0, stream>>>(ids, masks, emb, h32, h16);
  prep_w_kernel<<<(3 * DIM * DIM + 255) / 256, 256, 0, stream>>>(Wq, Wk, Wv, wt16);
  prep_cb_kernel<<<(KCB + 255) / 256, 256, 0, stream>>>(cb, cb16, cn);
  qkv_kernel<<<N_TOK / (16 * 8), 256, 0, stream>>>(h16, wt16, bq, bk, bv, q16, k16, vt16);
  attn_kernel<<<N_TOK / (16 * 8), 256, 0, stream>>>(q16, k16, vt16, ha32, ha16);
  vq_kernel<<<N_TOK / (16 * 8), 256, 0, stream>>>(ha16, cb16, cn, vidx);
  final_kernel<<<BATCH, DIM, 0, stream>>>(ha32, vidx, masks, cb, W_enc, b_enc, out);
}
